// UGFMEncoder_18287970747041
// MI455X (gfx1250) — matrix operand order repaired and checked by reading, not compiled, not run on hardware
//
#include <hip/hip_runtime.h>
#include <hip/hip_bf16.h>

typedef __attribute__((ext_vector_type(16))) _Float16 v16h;
typedef __attribute__((ext_vector_type(8)))  _Float16 v8h;
typedef __attribute__((ext_vector_type(8)))  float    v8f;

#define D_MODEL 128
#define NHEAD   8
#define DHEAD   16
#define ATT_SCALE 0.25f      // 1/sqrt(16)
#define LN_EPS  1e-5f

// ---------- helpers ----------
__device__ __forceinline__ unsigned fenc(float x){
  unsigned u = __float_as_uint(x);
  return (u & 0x80000000u) ? ~u : (u | 0x80000000u);   // order-preserving float->uint
}
__device__ __forceinline__ float fdec(unsigned u){
  return (u & 0x80000000u) ? __uint_as_float(u & 0x7fffffffu) : __uint_as_float(~u);
}

// A-matrix chunk load per ISA 16-bit A 16x32 layout:
// lane l holds row M=l%16; half i -> K = kk*32 + (i/8)*16 + (l>=16?8:0) + (i%8)
__device__ __forceinline__ v16h load_a_chunk(const _Float16* arow, int kk, int hiOff){
  union { v16h v; v8h h[2]; } u;
  u.h[0] = *(const v8h*)(arow + kk*32 + hiOff);
  u.h[1] = *(const v8h*)(arow + kk*32 + 16 + hiOff);
  return u.v;
}

// ---------- small prep kernels ----------
__global__ void cvt_f16_kernel(const float* __restrict__ in, _Float16* __restrict__ out, int n){
  int i = blockIdx.x*blockDim.x + threadIdx.x;
  if (i < n) out[i] = (_Float16)in[i];
}

// WnT[k][m][c] = Wn[k][c][m]  (so B K-dim is contiguous for the out-projection GEMM)
__global__ void transpose_w_kernel(const float* __restrict__ Wn, _Float16* __restrict__ WnT, int K){
  int idx = blockIdx.x*blockDim.x + threadIdx.x;
  int total = K * D_MODEL * D_MODEL;
  if (idx >= total) return;
  int k = idx >> 14;
  int r = idx & 16383;
  int m = r >> 7, c = r & 127;
  WnT[idx] = (_Float16)Wn[((size_t)k << 14) + ((size_t)c << 7) + m];
}

__global__ void gather_embed_kernel(const float* __restrict__ emb, const int* __restrict__ sid,
                                    float* __restrict__ feat0, _Float16* __restrict__ featH, int N){
  int idx = blockIdx.x*blockDim.x + threadIdx.x;
  if (idx >= N * D_MODEL) return;
  int n = idx >> 7, c = idx & 127;
  float v = emb[(size_t)sid[n]*D_MODEL + c];
  feat0[idx] = v;
  featH[idx] = (_Float16)v;
}

// ---------- bucketing by node key type ----------
__global__ void bucket_zero_kernel(int* counts, int* cursor, int K){
  int i = threadIdx.x;
  if (i < K){ counts[i] = 0; cursor[i] = 0; }
}
__global__ void bucket_count_kernel(const int* __restrict__ key, int* __restrict__ counts, int N){
  int i = blockIdx.x*blockDim.x + threadIdx.x;
  if (i < N) atomicAdd(&counts[key[i]], 1);
}
__global__ void bucket_scan_kernel(const int* __restrict__ counts, int* offsets, int* tileStart, int K){
  if (blockIdx.x == 0 && threadIdx.x == 0){
    int o = 0, t = 0;
    for (int k = 0; k < K; ++k){
      offsets[k] = o; tileStart[k] = t;
      o += counts[k];
      t += (counts[k] + 15) >> 4;
    }
    offsets[K] = o; tileStart[K] = t;
  }
}
__global__ void bucket_scatter_kernel(const int* __restrict__ key, const int* __restrict__ offsets,
                                      int* __restrict__ cursor, int* __restrict__ perm, int N){
  int i = blockIdx.x*blockDim.x + threadIdx.x;
  if (i >= N) return;
  int k = key[i];
  int p = atomicAdd(&cursor[k], 1);
  perm[offsets[k] + p] = i;
}

// ---------- WMMA GEMM: dense per-type K/V projections ----------
// C[n, c] = sum_m feat[n,m] * W[t][c][m]; B[K=m, N=c] read from W rows (contiguous in m).
__global__ void gemm_kv_kernel(const _Float16* __restrict__ featH,
                               const _Float16* __restrict__ WkH,
                               const _Float16* __restrict__ WvH,
                               _Float16* __restrict__ kall,
                               _Float16* __restrict__ vall,
                               int N, int T){
  int rowTile = blockIdx.x;
  int tz = (int)blockIdx.y;
  int t = (tz < T) ? tz : (tz - T);
  const _Float16* W = ((tz < T) ? WkH : WvH) + (size_t)t * D_MODEL * D_MODEL;
  _Float16* Out = ((tz < T) ? kall : vall) + (size_t)t * N * D_MODEL;

  int lane = threadIdx.x;
  int m = lane & 15, hi = lane >> 4;
  int row0 = rowTile * 16;
  const bool fullTile = (row0 + 16 <= N);       // wave-uniform

  int node = row0 + m; if (node >= N) node = N - 1;
  const _Float16* arow = featH + (size_t)node * D_MODEL;

  v16h A[4];
  #pragma unroll
  for (int kk = 0; kk < 4; ++kk) A[kk] = load_a_chunk(arow, kk, hi * 8);

  #pragma unroll
  for (int ct = 0; ct < 8; ++ct){
    const _Float16* bcol = W + (size_t)(ct*16 + m) * D_MODEL + hi*8;
    v8f acc = {};
    #pragma unroll
    for (int kk = 0; kk < 4; ++kk){
      v16h B = load_a_chunk(bcol, kk, 0);
      acc = __builtin_amdgcn_wmma_f32_16x16x32_f16(false, A[kk], false, B, (short)0, acc, false, false);
    }
    // C/D layout: VGPR r -> element (M = r + hi*8, Ncol = m)
    _Float16* obase = Out + (size_t)(row0 + hi*8) * D_MODEL + ct*16 + m;
    if (fullTile){
      #pragma unroll
      for (int r = 0; r < 8; ++r) obase[(size_t)r * D_MODEL] = (_Float16)acc[r];
    } else {
      #pragma unroll
      for (int r = 0; r < 8; ++r){
        int rr = row0 + r + hi*8;
        if (rr < N) Out[(size_t)rr * D_MODEL + ct*16 + m] = (_Float16)acc[r];
      }
    }
  }
}

// ---------- WMMA GEMM over per-type node buckets (Q projection / out projection) ----------
__global__ void gemm_bucketed_kernel(const _Float16* __restrict__ Ah, const _Float16* __restrict__ Wt,
                                     float* __restrict__ Out, const int* __restrict__ perm,
                                     const int* __restrict__ offsets, const int* __restrict__ tileStart,
                                     int N, int K){
  int tile = blockIdx.x;
  if (tile >= tileStart[K]) return;
  int k = 0;
  while (tile >= tileStart[k + 1]) k++;          // uniform scalar search over <=16 entries
  int lt = tile - tileStart[k];
  int rowBase = offsets[k] + lt * 16;
  int lim = offsets[k + 1];
  const bool fullTile = (rowBase + 16 <= lim);   // wave-uniform

  int lane = threadIdx.x;
  int m = lane & 15, hi = lane >> 4;
  int ridx = rowBase + m;
  int node = perm[(ridx < lim) ? ridx : rowBase];
  const _Float16* arow = Ah + (size_t)node * D_MODEL;

  v16h A[4];
  #pragma unroll
  for (int kk = 0; kk < 4; ++kk) A[kk] = load_a_chunk(arow, kk, hi * 8);

  // destination rows owned by this lane (M = r + hi*8)
  int nd[8];
  if (fullTile){
    #pragma unroll
    for (int r = 0; r < 8; ++r) nd[r] = perm[rowBase + r + hi*8];
  }

  const _Float16* W = Wt + (size_t)k * D_MODEL * D_MODEL;
  #pragma unroll
  for (int ct = 0; ct < 8; ++ct){
    const _Float16* bcol = W + (size_t)(ct*16 + m) * D_MODEL + hi*8;
    v8f acc = {};
    #pragma unroll
    for (int kk = 0; kk < 4; ++kk){
      v16h B = load_a_chunk(bcol, kk, 0);
      acc = __builtin_amdgcn_wmma_f32_16x16x32_f16(false, A[kk], false, B, (short)0, acc, false, false);
    }
    if (fullTile){
      #pragma unroll
      for (int r = 0; r < 8; ++r)
        Out[(size_t)nd[r] * D_MODEL + ct*16 + m] = acc[r];
    } else {
      #pragma unroll
      for (int r = 0; r < 8; ++r){
        int rr = rowBase + r + hi*8;
        if (rr < lim) Out[(size_t)perm[rr] * D_MODEL + ct*16 + m] = acc[r];
      }
    }
  }
}

// ---------- edge-wise attention ----------
__global__ void init_stats_kernel(unsigned* __restrict__ nodeMax, float* __restrict__ denom,
                                  float* __restrict__ agg, int N){
  int idx = blockIdx.x*blockDim.x + threadIdx.x;
  if (idx < N * D_MODEL) agg[idx] = 0.f;
  if (idx < N * NHEAD){ nodeMax[idx] = fenc(-1e30f); denom[idx] = 0.f; }
}

__global__ void edge_score_kernel(const float* __restrict__ q, const _Float16* __restrict__ kall,
                                  const int* __restrict__ src, const int* __restrict__ dst,
                                  const int* __restrict__ etype, float* __restrict__ score,
                                  unsigned* __restrict__ nodeMax, int E, int N){
  int idx = blockIdx.x*blockDim.x + threadIdx.x;
  if (idx >= E * NHEAD) return;
  int e = idx >> 3, h = idx & 7;
  int s = src[e], d = dst[e], t = etype[e];
  const float*   qr = q    + (size_t)d * D_MODEL + h * DHEAD;
  const _Float16* kr = kall + ((size_t)t * N + s) * D_MODEL + h * DHEAD;
  float acc = 0.f;
  #pragma unroll
  for (int i = 0; i < DHEAD; ++i) acc += qr[i] * (float)kr[i];
  acc *= ATT_SCALE;
  score[idx] = acc;
  atomicMax(&nodeMax[d * NHEAD + h], fenc(acc));
}

__global__ void edge_exp_kernel(float* __restrict__ score, const unsigned* __restrict__ nodeMax,
                                float* __restrict__ denom, const int* __restrict__ dst, int E){
  int idx = blockIdx.x*blockDim.x + threadIdx.x;
  if (idx >= E * NHEAD) return;
  int e = idx >> 3, h = idx & 7;
  int d = dst[e];
  float a = __expf(score[idx] - fdec(nodeMax[d * NHEAD + h]));
  score[idx] = a;
  atomicAdd(&denom[d * NHEAD + h], a);
}

__global__ void edge_agg_kernel(const float* __restrict__ score, const float* __restrict__ denom,
                                const _Float16* __restrict__ vall,
                                const int* __restrict__ src, const int* __restrict__ dst,
                                const int* __restrict__ etype, float* __restrict__ agg, int E, int N){
  long long idx = (long long)blockIdx.x * blockDim.x + threadIdx.x;
  if (idx >= (long long)E * D_MODEL) return;
  int e = (int)(idx >> 7);
  int c = (int)(idx & 127);
  int h = c >> 4;
  int d = dst[e];
  float attn = score[e * NHEAD + h] / fmaxf(denom[d * NHEAD + h], 1e-9f);
  int s = src[e], t = etype[e];
  float v = (float)vall[((size_t)t * N + s) * D_MODEL + c];
  atomicAdd(&agg[(size_t)d * D_MODEL + c], attn * v);
}

// ---------- ReLU + LayerNorm (+ optional residual), one wave per node ----------
__global__ void relu_ln_kernel(const float* __restrict__ in, const float* __restrict__ resid,
                               float* __restrict__ outF, _Float16* __restrict__ outH, int N){
  int gwave = (int)((blockIdx.x * blockDim.x + threadIdx.x) >> 5);
  int lane = threadIdx.x & 31;
  if (gwave >= N) return;
  const float* row = in + (size_t)gwave * D_MODEL;
  float x[4]; float s = 0.f;
  #pragma unroll
  for (int j = 0; j < 4; ++j){ float v = fmaxf(row[lane + 32*j], 0.f); x[j] = v; s += v; }
  #pragma unroll
  for (int o = 16; o > 0; o >>= 1) s += __shfl_xor(s, o, 32);
  float mu = s * (1.f / 128.f);
  float var = 0.f;
  #pragma unroll
  for (int j = 0; j < 4; ++j){ float d = x[j] - mu; var += d * d; }
  #pragma unroll
  for (int o = 16; o > 0; o >>= 1) var += __shfl_xor(var, o, 32);
  float inv = rsqrtf(var * (1.f / 128.f) + LN_EPS);
  #pragma unroll
  for (int j = 0; j < 4; ++j){
    float y = (x[j] - mu) * inv;
    size_t idx = (size_t)gwave * D_MODEL + lane + 32*j;
    if (resid) y += resid[idx];
    if (outF) outF[idx] = y;
    if (outH) outH[idx] = (_Float16)y;
  }
}

// ---------- host ----------
extern "C" void kernel_launch(void* const* d_in, const int* in_sizes, int n_in,
                              void* d_out, int out_size, void* d_ws, size_t ws_size,
                              hipStream_t stream){
  const float* emb = (const float*)d_in[0];
  const float* Wq  = (const float*)d_in[1];
  const float* Wk  = (const float*)d_in[2];
  const float* Wv  = (const float*)d_in[3];
  const float* Wn  = (const float*)d_in[4];
  const int* sid   = (const int*)d_in[5];
  const int* key   = (const int*)d_in[6];
  const int* et    = (const int*)d_in[7];
  const int* esrc  = (const int*)d_in[8];
  const int* edst  = (const int*)d_in[9];

  const int N = in_sizes[5];
  const int E = in_sizes[7];
  const int K = in_sizes[1] / (D_MODEL * D_MODEL);
  const int T = in_sizes[2] / (D_MODEL * D_MODEL);

  char* p = (char*)d_ws;
  auto alloc = [&](size_t bytes)->char*{
    char* r = p; p += (bytes + 255) & ~(size_t)255; return r;
  };
  float*    feat0  = (float*)   alloc((size_t)N * D_MODEL * 4);
  _Float16* featH  = (_Float16*)alloc((size_t)N * D_MODEL * 2);
  float*    qbuf   = (float*)   alloc((size_t)N * D_MODEL * 4);   // q, then out-projection
  _Float16* kall   = (_Float16*)alloc((size_t)T * N * D_MODEL * 2);
  _Float16* vall   = (_Float16*)alloc((size_t)T * N * D_MODEL * 2);
  float*    scoreA = (float*)   alloc((size_t)E * NHEAD * 4);
  unsigned* nodeMax= (unsigned*)alloc((size_t)N * NHEAD * 4);
  float*    denom  = (float*)   alloc((size_t)N * NHEAD * 4);
  float*    agg    = (float*)   alloc((size_t)N * D_MODEL * 4);
  _Float16* aggH   = (_Float16*)alloc((size_t)N * D_MODEL * 2);
  _Float16* WqH    = (_Float16*)alloc((size_t)K * D_MODEL * D_MODEL * 2);
  _Float16* WkH    = (_Float16*)alloc((size_t)T * D_MODEL * D_MODEL * 2);
  _Float16* WvH    = (_Float16*)alloc((size_t)T * D_MODEL * D_MODEL * 2);
  _Float16* WnT    = (_Float16*)alloc((size_t)K * D_MODEL * D_MODEL * 2);
  int* ints   = (int*)alloc(512);
  int* counts = ints, *cursor = ints + 32, *offsets = ints + 64, *tileStart = ints + 96;
  int* perm   = (int*)alloc((size_t)N * 4);

  dim3 b256(256);
  auto blocks = [](long long n, int bs){ return dim3((unsigned)((n + bs - 1) / bs)); };

  // weight conversion / transpose
  cvt_f16_kernel<<<blocks((long long)K*16384,256), b256, 0, stream>>>(Wq, WqH, K*16384);
  cvt_f16_kernel<<<blocks((long long)T*16384,256), b256, 0, stream>>>(Wk, WkH, T*16384);
  cvt_f16_kernel<<<blocks((long long)T*16384,256), b256, 0, stream>>>(Wv, WvH, T*16384);
  transpose_w_kernel<<<blocks((long long)K*16384,256), b256, 0, stream>>>(Wn, WnT, K);

  // bucket nodes by key type
  bucket_zero_kernel<<<1, 64, 0, stream>>>(counts, cursor, K);
  bucket_count_kernel<<<blocks(N,256), b256, 0, stream>>>(key, counts, N);
  bucket_scan_kernel<<<1, 32, 0, stream>>>(counts, offsets, tileStart, K);
  bucket_scatter_kernel<<<blocks(N,256), b256, 0, stream>>>(key, offsets, cursor, perm, N);

  // initial features
  gather_embed_kernel<<<blocks((long long)N*D_MODEL,256), b256, 0, stream>>>(emb, sid, feat0, featH, N);

  const int rowTiles = (N + 15) / 16;
  const int maxTiles = rowTiles + K;

  for (int conv = 0; conv < 2; ++conv){
    // Q projection per node type (bucketed WMMA GEMM)
    gemm_bucketed_kernel<<<dim3(maxTiles), dim3(32), 0, stream>>>(featH, WqH, qbuf, perm, offsets, tileStart, N, K);
    // K/V projections for all T edge types (dense WMMA GEMMs)
    gemm_kv_kernel<<<dim3(rowTiles, 2*T), dim3(32), 0, stream>>>(featH, WkH, WvH, kall, vall, N, T);
    // segment softmax + aggregation
    init_stats_kernel<<<blocks((long long)N*D_MODEL,256), b256, 0, stream>>>(nodeMax, denom, agg, N);
    edge_score_kernel<<<blocks((long long)E*NHEAD,256), b256, 0, stream>>>(qbuf, kall, esrc, edst, et, scoreA, nodeMax, E, N);
    edge_exp_kernel<<<blocks((long long)E*NHEAD,256), b256, 0, stream>>>(scoreA, nodeMax, denom, edst, E);
    edge_agg_kernel<<<blocks((long long)E*D_MODEL,256), b256, 0, stream>>>(scoreA, denom, vall, esrc, edst, et, agg, E, N);
    // out projection per node type (bucketed WMMA GEMM), reuse qbuf
    cvt_f16_kernel<<<blocks((long long)N*D_MODEL,256), b256, 0, stream>>>(agg, aggH, N*D_MODEL);
    gemm_bucketed_kernel<<<dim3(maxTiles), dim3(32), 0, stream>>>(aggH, WnT, qbuf, perm, offsets, tileStart, N, K);
    // relu + layernorm (+ block residual on final conv)
    if (conv == 0)
      relu_ln_kernel<<<blocks((long long)N*32,256), b256, 0, stream>>>(qbuf, nullptr, nullptr, featH, N);
    else
      relu_ln_kernel<<<blocks((long long)N*32,256), b256, 0, stream>>>(qbuf, feat0, (float*)d_out, nullptr, N);
  }
}
